// SAM_5102421148226
// MI455X (gfx1250) — matrix operand order repaired and checked by reading, compiled, not run on hardware
//
#include <hip/hip_runtime.h>
#include <hip/hip_bf16.h>

#define DEVI __device__ __forceinline__

typedef __attribute__((ext_vector_type(16))) _Float16 v16h;
typedef __attribute__((ext_vector_type(8)))  _Float16 v8h;
typedef __attribute__((ext_vector_type(8)))  float    v8f;

constexpr int Bsz  = 16;
constexpr int Cch  = 128;
constexpr int HW   = 1024;               // 32*32
constexpr int NPIX = Bsz * HW;           // 16384

DEVI int laneid() { return (int)(threadIdx.x & 31); }
// Wave-uniform wave id as an SGPR -> scalar branches, EXEC stays all-ones.
DEVI int waveid() { return __builtin_amdgcn_readfirstlane((int)(threadIdx.x >> 5)); }

// A-fragment (M=16 x K=32, f16, row-major, leading dim ld).
// Layout: lanes 0-15 -> M=lane, elems 0-15 = K 0..15;
//         lanes 16-31 -> M=lane-16, elems 0-15 = K 16..31.
DEVI v16h load_a16(const _Float16* base, int ld) {
  const int lane = laneid();
  const int row  = lane & 15;
  const int kb   = (lane >> 4) * 16;
  const _Float16* p = base + (size_t)row * ld + kb;
  v8h lo = *(const v8h*)p;
  v8h hi = *(const v8h*)(p + 8);
  v16h r;
#pragma unroll
  for (int i = 0; i < 8; ++i) { r[i] = lo[i]; r[i + 8] = hi[i]; }
  return r;
}

// B-fragment (K=32 x N=16, f16) loaded from a TRANSPOSED [N][K] buffer:
// lane n<16 holds col n, K=0..15 contiguous; lanes 16-31 hold K=16..31.
DEVI v16h load_b16(const _Float16* baseT, int ld) {
  const int lane = laneid();
  const int col  = lane & 15;
  const int kb   = (lane >> 4) * 16;
  return *(const v16h*)(baseT + (size_t)col * ld + kb);
}

// Per-lane address of this lane's slice of a B-fragment (for prefetch).
DEVI const _Float16* bfrag_lane_addr(const _Float16* baseT, int ld) {
  const int lane = laneid();
  return baseT + (size_t)(lane & 15) * ld + (lane >> 4) * 16;
}

DEVI v8f wmma_f16(v16h a, v16h b, v8f c) {
  return __builtin_amdgcn_wmma_f32_16x16x32_f16(false, a, false, b, (short)0, c,
                                                false, false);
}

// ---------------------------------------------------------------------------
// K0: [B][128][1024] f32  ->  [B][1024][128] f16   (LDS 32x32 tile transpose)
// ---------------------------------------------------------------------------
__global__ void __launch_bounds__(256)
transpose_kernel(const float* __restrict__ src, _Float16* __restrict__ dst) {
  __shared__ float tile[32][33];
  const int bid = blockIdx.x;
  const int hwt = bid & 31;
  const int ct  = (bid >> 5) & 3;
  const int b   = bid >> 7;
  const int c0 = ct * 32, hw0 = hwt * 32;
  const int tx = threadIdx.x & 31;
  const int ty = threadIdx.x >> 5;           // 0..7
#pragma unroll
  for (int i = 0; i < 4; ++i) {
    const int c = ty + i * 8;
    tile[c][tx] = src[((size_t)b * Cch + c0 + c) * HW + hw0 + tx];
  }
  __syncthreads();
#pragma unroll
  for (int i = 0; i < 4; ++i) {
    const int hw = ty + i * 8;
    dst[((size_t)b * HW + hw0 + hw) * Cch + c0 + tx] = (_Float16)tile[tx][hw];
  }
}

__global__ void __launch_bounds__(256)
cvt_kernel(const float* __restrict__ s, _Float16* __restrict__ d, int n) {
  for (int i = blockIdx.x * 256 + threadIdx.x; i < n; i += gridDim.x * 256)
    d[i] = (_Float16)s[i];
}

// ---------------------------------------------------------------------------
// K1: projections. One wave per (o-tile, 4 n-tiles): the weight A-fragments
// are loaded once and reused for 4 output tiles (16 WMMAs / wave).
// ---------------------------------------------------------------------------
__global__ void __launch_bounds__(128)
proj_kernel(const _Float16* __restrict__ hw16, const float* __restrict__ h_b,
            const _Float16* __restrict__ mw16, const float* __restrict__ m_b,
            const _Float16* __restrict__ ihT,  const _Float16* __restrict__ imT,
            _Float16* __restrict__ Qh, _Float16* __restrict__ Kh,
            _Float16* __restrict__ Vh, _Float16* __restrict__ Km,
            _Float16* __restrict__ Vm) {
  const int gw  = blockIdx.x * 4 + waveid();             // global wave id (SGPR)
  const int b   = gw / 640;                              // 40 o-tiles * 16 quads
  const int rem = gw % 640;
  const int ot  = rem / 16;                              // 0..23 h, 24..39 m
  const int nq  = rem % 16;                              // quad of n-tiles
  const int lane = laneid();
  const int n    = lane & 15;
  const int moff = (lane >> 4) * 8;

  const bool is_h = ot < 24;
  const int  oth  = is_h ? ot : ot - 24;
  const _Float16* W    = is_h ? hw16 : mw16;
  const float*    bias = is_h ? h_b : m_b;
  const _Float16* XT   = (is_h ? ihT : imT) + (size_t)b * HW * Cch;
  const int orow = oth * 16;
  const int sect = oth >> 3;   // h: 0=Q,1=K,2=V ; m: 0=Km,1=Vm

  v16h aw[4];
#pragma unroll
  for (int kk = 0; kk < 4; ++kk)
    aw[kk] = load_a16(W + (size_t)orow * Cch + kk * 32, Cch);

  float bv[8];
#pragma unroll
  for (int r = 0; r < 8; ++r) bv[r] = bias[orow + moff + r];

  for (int t = 0; t < 4; ++t) {
    const int nt = nq * 4 + t;
    v16h bf[4];
#pragma unroll
    for (int kk = 0; kk < 4; ++kk)
      bf[kk] = load_b16(XT + (size_t)nt * 16 * Cch + kk * 32, Cch);
    v8f acc = {};
#pragma unroll
    for (int kk = 0; kk < 4; ++kk) acc = wmma_f16(aw[kk], bf[kk], acc);
#pragma unroll
    for (int r = 0; r < 8; ++r) acc[r] += bv[r];

    const int pix = nt * 16 + n;
    if (is_h) {
      if (sect < 2) {            // pixel-major Q or K
        _Float16* dst = (sect == 0) ? Qh : Kh;
        v8h h;
#pragma unroll
        for (int r = 0; r < 8; ++r) h[r] = (_Float16)acc[r];
        *(v8h*)(dst + ((size_t)b * HW + pix) * Cch + (orow & 127) + moff) = h;
      } else {                   // channel-major V
        const int c0 = (orow - 256) + moff;
#pragma unroll
        for (int r = 0; r < 8; ++r)
          Vh[((size_t)b * Cch + c0 + r) * HW + pix] = (_Float16)acc[r];
      }
    } else {
      if (sect == 0) {           // pixel-major Km
        v8h h;
#pragma unroll
        for (int r = 0; r < 8; ++r) h[r] = (_Float16)acc[r];
        *(v8h*)(Km + ((size_t)b * HW + pix) * Cch + orow + moff) = h;
      } else {                   // channel-major Vm
        const int c0 = (orow - 128) + moff;
#pragma unroll
        for (int r = 0; r < 8; ++r)
          Vm[((size_t)b * Cch + c0 + r) * HW + pix] = (_Float16)acc[r];
      }
    }
  }
}

// ---------------------------------------------------------------------------
// K2: fused flash-attention. One workgroup (4 waves) per (batch, 16-row qtile).
// Score slab [16][1024] lives in LDS (32 KB); P row-major == B^T of V*P^T.
// ---------------------------------------------------------------------------
DEVI void attn_one(const v16h aq[4],
                   const _Float16* __restrict__ Kpix,   // [HW][128], batch base
                   const _Float16* __restrict__ Vcm,    // [128][HW], batch base
                   _Float16* P, float* red, float* rowm, float* rows,
                   _Float16* zdst, int zoff) {          // zdst = ZcatT+(b*HW+n0)*256
  const int lane = laneid();
  const int wid  = waveid();
  const int n    = lane & 15;
  const int moff = (lane >> 4) * 8;

  // ---- scores: S[16][1024] -> LDS (f16) ----
  for (int mt = wid; mt < 64; mt += 4) {
    if (mt + 4 < 64)   // prefetch next K-tile (global_prefetch_b8)
      __builtin_prefetch(bfrag_lane_addr(Kpix + (size_t)(mt + 4) * 16 * Cch, Cch),
                         0, 3);
    v16h bf[4];        // independent regs -> loads clause + single staged wait
#pragma unroll
    for (int kk = 0; kk < 4; ++kk)
      bf[kk] = load_b16(Kpix + (size_t)mt * 16 * Cch + kk * 32, Cch);
    v8f acc = {};
#pragma unroll
    for (int kk = 0; kk < 4; ++kk) acc = wmma_f16(aq[kk], bf[kk], acc);
    const int m = mt * 16 + n;
#pragma unroll
    for (int r = 0; r < 8; ++r) P[(size_t)(r + moff) * HW + m] = (_Float16)acc[r];
  }
  __syncthreads();

  // ---- softmax over each of 16 rows (8 threads / row) ----
  {
    const int row = threadIdx.x >> 3;
    const int seg = threadIdx.x & 7;
    const int base = row * HW + seg * 128;
    float mx = -3.0e38f;
    for (int j = 0; j < 128; ++j) mx = fmaxf(mx, (float)P[base + j]);
    red[row * 8 + seg] = mx;
    __syncthreads();
    if (seg == 0) {
      float v = red[row * 8];
      for (int j = 1; j < 8; ++j) v = fmaxf(v, red[row * 8 + j]);
      rowm[row] = v;
    }
    __syncthreads();
    const float rm = rowm[row];
    float s = 0.f;
    for (int j = 0; j < 128; ++j) {
      const float e = __expf((float)P[base + j] - rm);
      P[base + j] = (_Float16)e;
      s += e;
    }
    red[row * 8 + seg] = s;
    __syncthreads();
    if (seg == 0) {
      float v = 0.f;
      for (int j = 0; j < 8; ++j) v += red[row * 8 + j];
      rows[row] = 1.0f / v;
    }
    __syncthreads();
  }

  // ---- Z[c][n] = sum_m V[c][m] * P[n][m] ; scale by 1/rowsum ----
  for (int ct = wid * 2; ct < wid * 2 + 2; ++ct) {
    v8f acc = {};
    const _Float16* vb = Vcm + (size_t)ct * 16 * HW;
    for (int kk0 = 0; kk0 < 32; kk0 += 4) {
      if (kk0 + 4 < 32)   // prefetch this lane's next V A-slice group
        __builtin_prefetch(vb + (size_t)(lane & 15) * HW + (kk0 + 4) * 32 +
                               (lane >> 4) * 8, 0, 3);
      v16h av[4], bv[4];
#pragma unroll
      for (int j = 0; j < 4; ++j)
        av[j] = load_a16(vb + (kk0 + j) * 32, HW);
#pragma unroll
      for (int j = 0; j < 4; ++j)
        bv[j] = load_b16(P + (kk0 + j) * 32, HW);
#pragma unroll
      for (int j = 0; j < 4; ++j) acc = wmma_f16(av[j], bv[j], acc);
    }
    const float sc = rows[n];
    v8h h;
#pragma unroll
    for (int r = 0; r < 8; ++r) h[r] = (_Float16)(acc[r] * sc);
    *(v8h*)(zdst + (size_t)n * 256 + zoff + ct * 16 + moff) = h;
  }
  __syncthreads();     // P reused by the next attention half
}

__global__ void __launch_bounds__(128)
attn_kernel(const _Float16* __restrict__ Qh, const _Float16* __restrict__ Kh,
            const _Float16* __restrict__ Vh, const _Float16* __restrict__ Km,
            const _Float16* __restrict__ Vm, _Float16* __restrict__ ZcatT) {
  __shared__ __align__(32) _Float16 P[16 * HW];   // 32 KB score/prob slab
  __shared__ float red[16 * 8];
  __shared__ float rowm[16];
  __shared__ float rows[16];

  const int b  = blockIdx.x >> 6;
  const int qt = blockIdx.x & 63;
  const int n0 = qt * 16;

  const _Float16* qbase = Qh + ((size_t)b * HW + n0) * Cch;
  v16h aq[4];
#pragma unroll
  for (int kk = 0; kk < 4; ++kk) aq[kk] = load_a16(qbase + kk * 32, Cch);

  _Float16* zdst = ZcatT + ((size_t)b * HW + n0) * 256;
  attn_one(aq, Kh + (size_t)b * HW * Cch, Vh + (size_t)b * Cch * HW,
           P, red, rowm, rows, zdst, 0);
  attn_one(aq, Km + (size_t)b * HW * Cch, Vm + (size_t)b * Cch * HW,
           P, red, rowm, rows, zdst, 128);
}

// ---------------------------------------------------------------------------
// K3: Z = z1_w @ Zcat (K=256), gates = z2_w @ [Z; ih] (K=256), then gating.
// ---------------------------------------------------------------------------
__global__ void __launch_bounds__(128)
out_kernel(const _Float16* __restrict__ z1w, const float* __restrict__ z1b,
           const _Float16* __restrict__ z2w, const float* __restrict__ z2b,
           const _Float16* __restrict__ ZcatT, const _Float16* __restrict__ ihT,
           const float* __restrict__ im, float* __restrict__ oh,
           float* __restrict__ om) {
  __shared__ __align__(32) _Float16 Zt[16 * 128];  // [pixel][c] f16
  __shared__ float G[384 * 16];                    // gates [o][pixel]
  const int p0   = blockIdx.x * 16;
  const int lane = laneid();
  const int wid  = waveid();
  const int n    = lane & 15;
  const int moff = (lane >> 4) * 8;
  const _Float16* zrow = ZcatT + (size_t)p0 * 256;

  for (int ot = wid * 2; ot < wid * 2 + 2; ++ot) {    // 8 o-tiles of Z
    v8f acc = {};
#pragma unroll
    for (int g = 0; g < 2; ++g) {                     // 2 groups of 4 k-steps
      v16h af[4], bf[4];
#pragma unroll
      for (int j = 0; j < 4; ++j) {
        const int kk = g * 4 + j;
        af[j] = load_a16(z1w + (size_t)ot * 16 * 256 + kk * 32, 256);
        bf[j] = load_b16(zrow + kk * 32, 256);
      }
#pragma unroll
      for (int j = 0; j < 4; ++j) acc = wmma_f16(af[j], bf[j], acc);
    }
    v8h h;
#pragma unroll
    for (int r = 0; r < 8; ++r) h[r] = (_Float16)(acc[r] + z1b[ot * 16 + moff + r]);
    *(v8h*)(&Zt[n * 128 + ot * 16 + moff]) = h;
  }
  __syncthreads();

  for (int ot = wid * 6; ot < wid * 6 + 6; ++ot) {    // 24 o-tiles of gates
    v8f acc = {};
    {
      v16h af[4], bf[4];                              // K 0..127 from Z (LDS)
#pragma unroll
      for (int kk = 0; kk < 4; ++kk) {
        af[kk] = load_a16(z2w + (size_t)ot * 16 * 256 + kk * 32, 256);
        bf[kk] = load_b16(&Zt[kk * 32], 128);
      }
#pragma unroll
      for (int kk = 0; kk < 4; ++kk) acc = wmma_f16(af[kk], bf[kk], acc);
    }
    {
      v16h af[4], bf[4];                              // K 128..255 from ih^T
#pragma unroll
      for (int kk = 0; kk < 4; ++kk) {
        af[kk] = load_a16(z2w + (size_t)ot * 16 * 256 + 128 + kk * 32, 256);
        bf[kk] = load_b16(ihT + (size_t)p0 * 128 + kk * 32, 128);
      }
#pragma unroll
      for (int kk = 0; kk < 4; ++kk) acc = wmma_f16(af[kk], bf[kk], acc);
    }
#pragma unroll
    for (int r = 0; r < 8; ++r)
      G[(ot * 16 + moff + r) * 16 + n] = acc[r] + z2b[ot * 16 + moff + r];
  }
  __syncthreads();

  for (int e = threadIdx.x; e < 2048; e += 128) {     // 128 c x 16 pixels
    const int c  = e >> 4;
    const int nn = e & 15;
    const float ov = G[c * 16 + nn];
    const float gv = G[(128 + c) * 16 + nn];
    const float iv = G[(256 + c) * 16 + nn];
    const int p  = p0 + nn;
    const int b  = p >> 10;
    const int hw = p & 1023;
    const size_t idx = (size_t)b * Cch * HW + (size_t)c * HW + hw;
    const float ig  = 1.f / (1.f + __expf(-iv));
    const float omv = tanhf(gv) * ig + (1.f - ig) * im[idx];
    const float ohv = (1.f / (1.f + __expf(-ov))) * omv;
    om[idx] = omv;
    oh[idx] = ohv;
  }
}

// ---------------------------------------------------------------------------
extern "C" void kernel_launch(void* const* d_in, const int* in_sizes, int n_in,
                              void* d_out, int out_size, void* d_ws, size_t ws_size,
                              hipStream_t stream) {
  (void)in_sizes; (void)n_in; (void)out_size; (void)ws_size;
  const float* ih  = (const float*)d_in[0];
  const float* im  = (const float*)d_in[1];
  const float* h_w = (const float*)d_in[2];
  const float* h_b = (const float*)d_in[3];
  const float* m_w = (const float*)d_in[4];
  const float* m_b = (const float*)d_in[5];
  const float* z1w = (const float*)d_in[6];
  const float* z1b = (const float*)d_in[7];
  const float* z2w = (const float*)d_in[8];
  const float* z2b = (const float*)d_in[9];

  char* wsb = (char*)d_ws;
  size_t off = 0;
  auto carve = [&](size_t bytes) -> char* {
    char* p = wsb + off;
    off = (off + bytes + 255) & ~(size_t)255;
    return p;
  };
  _Float16* ihT   = (_Float16*)carve((size_t)NPIX * 128 * 2);
  _Float16* imT   = (_Float16*)carve((size_t)NPIX * 128 * 2);
  _Float16* hw16  = (_Float16*)carve((size_t)384 * 128 * 2);
  _Float16* mw16  = (_Float16*)carve((size_t)256 * 128 * 2);
  _Float16* z1w16 = (_Float16*)carve((size_t)128 * 256 * 2);
  _Float16* z2w16 = (_Float16*)carve((size_t)384 * 256 * 2);
  _Float16* Qh    = (_Float16*)carve((size_t)NPIX * 128 * 2);
  _Float16* Kh    = (_Float16*)carve((size_t)NPIX * 128 * 2);
  _Float16* Vh    = (_Float16*)carve((size_t)NPIX * 128 * 2);
  _Float16* Km    = (_Float16*)carve((size_t)NPIX * 128 * 2);
  _Float16* Vm    = (_Float16*)carve((size_t)NPIX * 128 * 2);
  _Float16* ZcatT = (_Float16*)carve((size_t)NPIX * 256 * 2);

  float* oh = (float*)d_out;
  float* om = oh + (size_t)Bsz * Cch * HW;

  transpose_kernel<<<2048, 256, 0, stream>>>(ih, ihT);
  transpose_kernel<<<2048, 256, 0, stream>>>(im, imT);
  cvt_kernel<<<(384 * 128 + 255) / 256, 256, 0, stream>>>(h_w, hw16, 384 * 128);
  cvt_kernel<<<(256 * 128 + 255) / 256, 256, 0, stream>>>(m_w, mw16, 256 * 128);
  cvt_kernel<<<(128 * 256 + 255) / 256, 256, 0, stream>>>(z1w, z1w16, 128 * 256);
  cvt_kernel<<<(384 * 256 + 255) / 256, 256, 0, stream>>>(z2w, z2w16, 384 * 256);

  proj_kernel<<<2560, 128, 0, stream>>>(hw16, h_b, mw16, m_b, ihT, imT,
                                        Qh, Kh, Vh, Km, Vm);
  attn_kernel<<<1024, 128, 0, stream>>>(Qh, Kh, Vh, Km, Vm, ZcatT);
  out_kernel<<<1024, 128, 0, stream>>>(z1w16, z1b, z2w16, z2b, ZcatT, ihT,
                                       im, oh, om);
}
